// ConceptNetEncoder_46763603919316
// MI455X (gfx1250) — matrix operand order repaired and checked by reading, not compiled, not run on hardware
//
#include <hip/hip_runtime.h>
#include <hip/hip_bf16.h>
#include <math.h>

#define VOCAB 100000
#define TOPK  20
#define EMB   300
#define BS    64
#define SEQ   256
#define NTOK  (BS*SEQ)

#define DP 320                 // K dim (emb dim) padded to 10*32
#define EP 304                 // N dim (cols of a) padded to 19*16
#define ROWS_PER_WG 128        // vocab rows per workgroup (8 waves x 16-row tiles)
#define KT (DP/32)             // 10 WMMA K-steps
#define NT (EP/16)             // 19 N tiles

typedef _Float16 v16h __attribute__((ext_vector_type(16)));
typedef float    v8f  __attribute__((ext_vector_type(8)));
typedef unsigned int u32x4 __attribute__((ext_vector_type(4)));
typedef int          i32x4 __attribute__((ext_vector_type(4)));
typedef int          i32x8 __attribute__((ext_vector_type(8)));

union Frag16 { uint4 u[2]; v16h v; };

// workspace layout (bytes)
#define WS_AT 0                         // f16 aT[EP][DP]  = 194,560 B
#define WS_B  (EP*DP*2)                 // f32 bpad[EP]    = 1,216 B
#define WS_SC 195840                    // f32 score[VOCAB]= 400,000 B (256-aligned)

#define A_BYTES (EP*DP*2)               // 194,560
#define A_ELEM8 (A_BYTES/8)             // 24,320 8-byte elements

#if __has_builtin(__builtin_amdgcn_tensor_load_to_lds) && __has_builtin(__builtin_amdgcn_s_wait_tensorcnt)
#define USE_TDM 1
#else
#define USE_TDM 0
#endif

// native CDNA5 tanh (single TRANS op, co-executes with WMMA)
__device__ __forceinline__ float fast_tanh(float x) {
#if __has_builtin(__builtin_amdgcn_tanhf)
    return __builtin_amdgcn_tanhf(x);
#else
    float r;
    asm("v_tanh_f32 %0, %1" : "=v"(r) : "v"(x));
    return r;
#endif
}

// ---------------------------------------------------------------------------
// Kernel 0: a[300][300] f32 -> aT[EP][DP] f16 (transposed, zero-padded);
//           b[300] -> bpad[EP] f32 (zero-padded)
// ---------------------------------------------------------------------------
__global__ void __launch_bounds__(256)
prep_kernel(const float* __restrict__ a, const float* __restrict__ b,
            _Float16* __restrict__ aT, float* __restrict__ bpad) {
    int gid = blockIdx.x * 256 + threadIdx.x;
    if (gid < EP * DP) {
        int n = gid / DP, k = gid % DP;
        float v = (n < EMB && k < EMB) ? a[k * EMB + n] : 0.f;
        aT[gid] = (_Float16)v;
    }
    if (gid < EP) bpad[gid] = (gid < EMB) ? b[gid] : 0.f;
}

// ---------------------------------------------------------------------------
// Kernel 1: per-vocab-row attention score:
//   score[v] = sum_n tanh( sum_d emb[v][d] * a[d][n] ) * b[n]
// Dense GEMM via v_wmma_f32_16x16x32_f16; aT + emb tile + bpad in LDS.
// ---------------------------------------------------------------------------
__global__ void __launch_bounds__(256)
score_kernel(const float* __restrict__ emb,
             const _Float16* __restrict__ aT,
             const float* __restrict__ bpad,
             float* __restrict__ score) {
    extern __shared__ _Float16 smem[];
    _Float16* aLds = smem;                          // [EP][DP]          194,560 B
    _Float16* eLds = smem + EP * DP;                // [ROWS_PER_WG][DP]  81,920 B
    float*    bLds = (float*)(smem + EP * DP + ROWS_PER_WG * DP); // [EP] 1,216 B

    const int tid = threadIdx.x;
    const int rowBase = blockIdx.x * ROWS_PER_WG;

#if USE_TDM
    // Tensor Data Mover: async DMA aT (flat 194,560 B, as 24,320 x 8B) -> LDS.
    if (tid < 32) {
        unsigned int lds = (unsigned int)(uintptr_t)(void*)aLds;
        unsigned long long ga = (unsigned long long)(uintptr_t)aT;
        u32x4 g0;
        g0[0] = 1u;                                        // count=1 (valid user D#)
        g0[1] = lds;                                       // lds_addr
        g0[2] = (unsigned int)(ga & 0xFFFFFFFFu);          // global_addr[31:0]
        g0[3] = (unsigned int)((ga >> 32) & 0x01FFFFFFu)   // global_addr[56:32]
              | (2u << 30);                                // type=2 ("image")
        i32x8 g1;
        g1[0] = (int)(3u << 16);                           // data_size=8B
        g1[1] = (int)((A_ELEM8 & 0xFFFFu) << 16);          // tensor_dim0[15:0]
        g1[2] = (int)((A_ELEM8 >> 16) | (1u << 16));       // tensor_dim0[31:16] | tensor_dim1=1
        g1[3] = (int)((unsigned)A_ELEM8 << 16);            // tile_dim0 = 24,320
        g1[4] = 1;                                         // tile_dim1=1, tile_dim2=0
        g1[5] = A_ELEM8;                                   // tensor_dim0_stride[31:0]
        g1[6] = 0;
        g1[7] = 0;
        i32x4 z4 = {0, 0, 0, 0};
#if __clang_major__ >= 23
        i32x8 z8 = {0, 0, 0, 0, 0, 0, 0, 0};
        __builtin_amdgcn_tensor_load_to_lds(g0, g1, z4, z4, z8, 0);
#else
        __builtin_amdgcn_tensor_load_to_lds(g0, g1, z4, z4, 0);
#endif
        __builtin_amdgcn_s_wait_tensorcnt((short)0);
    }
#else
    {   // fallback: cooperative flat b128 copy of aT
        const uint4* src = (const uint4*)aT;
        uint4* dst = (uint4*)aLds;
        const int n4 = A_BYTES / 16;
        for (int i = tid; i < n4; i += 256) dst[i] = src[i];
    }
#endif

    // cooperative fill: emb rows f32 -> f16, zero-pad k in [300,320)
    {
        const int total = ROWS_PER_WG * DP / 2;   // pairs of halfs
        for (int i = tid; i < total; i += 256) {
            int r = (i * 2) / DP;
            int k = (i * 2) % DP;
            int grow = rowBase + r; if (grow >= VOCAB) grow = VOCAB - 1;
            const float* er = emb + (size_t)grow * EMB;
            float f0 = (k     < EMB) ? er[k]     : 0.f;
            float f1 = (k + 1 < EMB) ? er[k + 1] : 0.f;
            eLds[r * DP + k]     = (_Float16)f0;
            eLds[r * DP + k + 1] = (_Float16)f1;
        }
    }
    for (int i = tid; i < EP; i += 256) bLds[i] = bpad[i];
    __syncthreads();

    const int wave = tid >> 5;
    const int lane = tid & 31;
    const int hi   = lane >> 4;    // half-wave select
    const int l16  = lane & 15;

    // A fragments (16x32 f16, ISA layout: lane<16 holds K runs {0..7,16..23},
    // lane>=16 holds {8..15,24..31}); preload all 10 K-steps for this wave's M tile.
    const _Float16* arow = eLds + (wave * 16 + l16) * DP;
    Frag16 afrag[KT];
    #pragma unroll
    for (int kt = 0; kt < KT; ++kt) {
        const _Float16* p = arow + kt * 32 + hi * 8;
        afrag[kt].u[0] = *(const uint4*)(p);
        afrag[kt].u[1] = *(const uint4*)(p + 16);
    }

    float partial[8] = {0.f,0.f,0.f,0.f,0.f,0.f,0.f,0.f};

    for (int nt = 0; nt < NT; ++nt) {
        // Preload ALL B fragments for this N tile so the ds_loads pipeline ahead
        // of the WMMA chain instead of a per-step s_wait_dscnt 0.
        const _Float16* brow = aLds + (nt * 16 + l16) * DP + hi * 8;
        Frag16 bf[KT];
        #pragma unroll
        for (int kt = 0; kt < KT; ++kt) {
            bf[kt].u[0] = *(const uint4*)(brow + kt * 32);
            bf[kt].u[1] = *(const uint4*)(brow + kt * 32 + 16);
        }
        v8f c = {};
        #pragma unroll
        for (int kt = 0; kt < KT; ++kt)
            c = __builtin_amdgcn_wmma_f32_16x16x32_f16(
                    false, afrag[kt].v, false, bf[kt].v,
                    (short)0, c, false, false);
        // epilogue: score contribution tanh(C[m][n]) * b[n] (native v_tanh_f32)
        float bb = bLds[nt * 16 + l16];
        #pragma unroll
        for (int v = 0; v < 8; ++v) partial[v] += fast_tanh(c[v]) * bb;
    }

    // C layout: VGPR v, lanes 0-15 -> row M=v, lanes 16-31 -> row M=v+8;
    // reduce the 16 lanes (distinct N columns) sharing each row.
    #pragma unroll
    for (int v = 0; v < 8; ++v) {
        float p = partial[v];
        p += __shfl_xor(p, 1, 32);
        p += __shfl_xor(p, 2, 32);
        p += __shfl_xor(p, 4, 32);
        p += __shfl_xor(p, 8, 32);
        partial[v] = p;
    }
    if (l16 == 0) {
        #pragma unroll
        for (int v = 0; v < 8; ++v) {
            int row = rowBase + wave * 16 + v + hi * 8;
            if (row < VOCAB) score[row] = partial[v];
        }
    }
}

// ---------------------------------------------------------------------------
// Kernel 2: one wave per token: gather 20 neighbor scores, softmax over K=20,
// weighted sum of the 20 f32 embedding rows (L2-resident gathers).
// ---------------------------------------------------------------------------
__global__ void __launch_bounds__(256)
attend_kernel(const int* __restrict__ ids,
              const int* __restrict__ neighbors,
              const float* __restrict__ emb,
              const float* __restrict__ score,
              float* __restrict__ out) {
    const int wave = threadIdx.x >> 5;
    const int lane = threadIdx.x & 31;
    const int t = blockIdx.x * 8 + wave;   // grid sized exactly: wave-uniform

    const int id = ids[t];
    int nid = 0;
    float e = -INFINITY;
    if (lane < TOPK) {
        nid = neighbors[id * TOPK + lane];
        e = score[nid];
    }
    float m = e;
    #pragma unroll
    for (int off = 16; off >= 1; off >>= 1) m = fmaxf(m, __shfl_xor(m, off, 32));
    float p = (lane < TOPK) ? __expf(e - m) : 0.f;
    float s = p;
    #pragma unroll
    for (int off = 16; off >= 1; off >>= 1) s += __shfl_xor(s, off, 32);
    const float inv = 1.f / s;

    float acc[10];
    #pragma unroll
    for (int j = 0; j < 10; ++j) acc[j] = 0.f;

    for (int k = 0; k < TOPK; ++k) {
        float ak = __shfl(p, k, 32) * inv;
        int   ik = __shfl(nid, k, 32);
        const float* row = emb + (size_t)ik * EMB;
        #pragma unroll
        for (int j = 0; j < 10; ++j) {
            int d = lane + 32 * j;
            if (d < EMB) acc[j] += ak * row[d];   // coalesced across the wave
        }
    }
    float* o = out + (size_t)t * EMB;
    #pragma unroll
    for (int j = 0; j < 10; ++j) {
        int d = lane + 32 * j;
        if (d < EMB) o[d] = acc[j];
    }
}

// ---------------------------------------------------------------------------
extern "C" void kernel_launch(void* const* d_in, const int* in_sizes, int n_in,
                              void* d_out, int out_size, void* d_ws, size_t ws_size,
                              hipStream_t stream) {
    const int*   ids       = (const int*)  d_in[0];  // [64,256]
    const int*   neighbors = (const int*)  d_in[1];  // [100000,20]
    const float* emb       = (const float*)d_in[2];  // [100000,300]
    const float* a         = (const float*)d_in[3];  // [300,300]
    const float* b         = (const float*)d_in[4];  // [300,1]
    float* out = (float*)d_out;                      // [64,256,300]

    char* ws = (char*)d_ws;
    _Float16* aT   = (_Float16*)(ws + WS_AT);
    float*    bpad = (float*)   (ws + WS_B);
    float*    sc   = (float*)   (ws + WS_SC);

    prep_kernel<<<(EP * DP + 255) / 256, 256, 0, stream>>>(a, b, aT, bpad);

    const int nblk = (VOCAB + ROWS_PER_WG - 1) / ROWS_PER_WG;   // 782
    const size_t shmem = (size_t)(EP * DP + ROWS_PER_WG * DP) * sizeof(_Float16)
                       + EP * sizeof(float);                    // 277,696 B
    score_kernel<<<nblk, 256, shmem, stream>>>(emb, aT, bpad, sc);

    attend_kernel<<<NTOK / 8, 256, 0, stream>>>(ids, neighbors, emb, sc, out);
}
